// TransformerBlock_13855564497335
// MI455X (gfx1250) — matrix operand order repaired and checked by reading, not compiled, not run on hardware
//
#include <hip/hip_runtime.h>
#include <hip/hip_bf16.h>

#define BATCH 256
#define TOK   361
#define CH    512
#define NH    8
#define DHEAD 64
#define MLPD  2048
#define BT    (BATCH * TOK)      // 92416 = 32 * 2888, 2888 % 8 == 0
#define TPAD  384
#define BH    (BATCH * NH)       // 2048

typedef __attribute__((ext_vector_type(16))) __bf16 v16bf;
typedef __attribute__((ext_vector_type(8)))  float  v8f;
typedef int v4i __attribute__((vector_size(16)));
typedef __attribute__((address_space(1))) v4i* g_v4i_ptr;   // global AS pointer to int4
typedef __attribute__((address_space(3))) v4i* l_v4i_ptr;   // LDS AS pointer to int4

union Frag { v16bf v; uint4 q[2]; };

__device__ __forceinline__ v8f wmma_bf16(v16bf a, v16bf b, v8f c) {
  return __builtin_amdgcn_wmma_f32_16x16x32_bf16(false, a, false, b, (short)0, c, false, false);
}

#if defined(__AMDGCN__) && __has_builtin(__builtin_amdgcn_global_load_async_to_lds_b128)
#define GEMM_ASYNC 1
#else
#define GEMM_ASYNC 0
#endif

__device__ __forceinline__ void wait_async_le1() {
#if GEMM_ASYNC
#if __has_builtin(__builtin_amdgcn_s_wait_asynccnt)
  __builtin_amdgcn_s_wait_asynccnt(1);
#else
  asm volatile("s_wait_asynccnt 0x1" ::: "memory");
#endif
#endif
}
__device__ __forceinline__ void wait_async_le0() {
#if GEMM_ASYNC
#if __has_builtin(__builtin_amdgcn_s_wait_asynccnt)
  __builtin_amdgcn_s_wait_asynccnt(0);
#else
  asm volatile("s_wait_asynccnt 0x0" ::: "memory");
#endif
#endif
}

// ---------------- weight convert + transpose:  W[K,N] f32 -> Wt[N,K] bf16 ----
__global__ __launch_bounds__(256) void cvt_w(const float* __restrict__ W,
                                             __bf16* __restrict__ Wt, int K, int N) {
  int i = blockIdx.x * blockDim.x + threadIdx.x;
  if (i >= K * N) return;
  int k = i / N, n = i % N;
  Wt[(size_t)n * K + k] = (__bf16)W[i];
}

__global__ void pack_bias(const float* __restrict__ bq, const float* __restrict__ bk,
                          const float* __restrict__ bv, float* __restrict__ dst) {
  int i = blockIdx.x * blockDim.x + threadIdx.x;
  if (i < 512)       dst[i] = bq[i];
  else if (i < 1024) dst[i] = bk[i - 512];
  else if (i < 1536) dst[i] = bv[i - 1024];
}

// ---------------- LayerNorm: wave per token, bf16 out ------------------------
__global__ __launch_bounds__(256) void ln_kernel(const float* __restrict__ x,
    const float* __restrict__ sc, const float* __restrict__ bi,
    __bf16* __restrict__ y, int ntok)
{
  int wave = blockIdx.x * (blockDim.x >> 5) + (threadIdx.x >> 5);
  int lane = threadIdx.x & 31;
  if (wave >= ntok) return;
  const float* xr = x + (size_t)wave * CH + lane * 16;
  float v[16];
  float s = 0.f;
#pragma unroll
  for (int i = 0; i < 16; i += 4) {
    float4 f = *(const float4*)(xr + i);
    v[i] = f.x; v[i + 1] = f.y; v[i + 2] = f.z; v[i + 3] = f.w;
    s += f.x + f.y + f.z + f.w;
  }
#pragma unroll
  for (int off = 16; off; off >>= 1) s += __shfl_xor(s, off, 32);
  float mu = s * (1.f / CH);
  float vs = 0.f;
#pragma unroll
  for (int i = 0; i < 16; ++i) { float d = v[i] - mu; vs += d * d; }
#pragma unroll
  for (int off = 16; off; off >>= 1) vs += __shfl_xor(vs, off, 32);
  float inv = rsqrtf(vs * (1.f / CH) + 1e-6f);
  union { __bf16 h[16]; uint4 q[2]; } o;
#pragma unroll
  for (int i = 0; i < 16; ++i) {
    int c = lane * 16 + i;
    o.h[i] = (__bf16)((v[i] - mu) * inv * sc[c] + bi[c]);
  }
  uint4* yp = (uint4*)(y + (size_t)wave * CH + lane * 16);
  yp[0] = o.q[0]; yp[1] = o.q[1];
}

// ---------------- bf16 WMMA GEMM: wave -> 32x64 tile, LDS-staged weights -----
// A [M,K] bf16 row-major, Bt [N,K] bf16 (B transposed), f32 accumulate.
// Requires: M % 32 == 0, (M/32) % 8 == 0, N % 64 == 0, K % 32 == 0, grid exact.
// flags bit0 = tanh-GELU.
__global__ __launch_bounds__(256) void gemm_bf16(
    const __bf16* __restrict__ A, const __bf16* __restrict__ Bt,
    const float* __restrict__ bias, const float* __restrict__ residual,
    float* __restrict__ outF, __bf16* __restrict__ outB,
    int M, int N, int K, int flags)
{
  __shared__ __align__(16) __bf16 bsh[2][64][32];   // double-buffered 64x32 weight tile
  int wid  = threadIdx.x >> 5;
  int lane = threadIdx.x & 31;
  int Mt = M >> 5;                                  // 32-row tiles; Mt % 8 == 0
  int wave = blockIdx.x * 8 + wid;
  int mt = wave % Mt, nt = wave / Mt;               // all 8 waves of a block: same nt
  int m0 = mt << 5, n0 = nt << 6;

  int ln15 = lane & 15;
  int kb8  = (lane >> 4) * 8;
  int kb16 = (lane >> 4) * 16;
  const __bf16* arow0 = A + (size_t)(m0 + ln15) * K;
  const __bf16* arow1 = A + (size_t)(m0 + 16 + ln15) * K;

  // block-cooperative weight staging: thread -> (row 0..63, 16B quad 0..3)
  int srow  = threadIdx.x >> 2;
  int squad = threadIdx.x & 3;
  const __bf16* sbase = Bt + (size_t)(n0 + srow) * K + squad * 8;

  auto stage = [&](int nb, int kc) {
#if GEMM_ASYNC
    __builtin_amdgcn_global_load_async_to_lds_b128(
        (g_v4i_ptr)(sbase + kc),
        (l_v4i_ptr)&bsh[nb][srow][squad * 8],
        0, 0);
#else
    *(uint4*)&bsh[nb][srow][squad * 8] = *(const uint4*)(sbase + kc);
#endif
  };

  v8f acc0[4] = {}, acc1[4] = {};
  int nCk = K >> 5;

  stage(0, 0);
  for (int ck = 0; ck < nCk; ++ck) {
    int cur = ck & 1;
    if (ck + 1 < nCk) {
      stage(cur ^ 1, (ck + 1) << 5);
      wait_async_le1();
    } else {
      wait_async_le0();
    }
    __syncthreads();
    int kc = ck << 5;
    __builtin_prefetch(arow0 + kc + 64, 0, 0);      // next A chunks -> global_prefetch
    __builtin_prefetch(arow1 + kc + 64, 0, 0);
    Frag a0, a1;
    a0.q[0] = *(const uint4*)(arow0 + kc + kb8);
    a0.q[1] = *(const uint4*)(arow0 + kc + kb8 + 16);
    a1.q[0] = *(const uint4*)(arow1 + kc + kb8);
    a1.q[1] = *(const uint4*)(arow1 + kc + kb8 + 16);
#pragma unroll
    for (int j = 0; j < 4; ++j) {
      const uint4* bp = (const uint4*)&bsh[cur][j * 16 + ln15][kb8];
      Frag b;
      b.q[0] = bp[0];
      b.q[1] = bp[2];
      acc0[j] = wmma_bf16(a0.v, b.v, acc0[j]);
      acc1[j] = wmma_bf16(a1.v, b.v, acc1[j]);
    }
    __syncthreads();
  }

  int mb = (lane >> 4) * 8;
#pragma unroll
  for (int half = 0; half < 2; ++half) {
    v8f* acc = half ? acc1 : acc0;
    int mh = m0 + half * 16 + mb;
#pragma unroll
    for (int j = 0; j < 4; ++j) {
      int n = n0 + j * 16 + ln15;
      float bv = bias ? bias[n] : 0.f;
#pragma unroll
      for (int i = 0; i < 8; ++i) {
        float val = acc[j][i] + bv;
        if (flags & 1) {
          float u = val;
          float t = 0.7978845608028654f * (u + 0.044715f * u * u * u);
          val = 0.5f * u * (1.f + tanhf(t));
        }
        size_t idx = (size_t)(mh + i) * N + n;
        if (residual) val += residual[idx];
        if (outF) outF[idx] = val;
        if (outB) outB[idx] = (__bf16)val;
      }
    }
  }
}

// ---------------- qkv [BT,1536] bf16 -> q,k [BH,TPAD,64], vT [BH,64,TPAD] ----
__global__ __launch_bounds__(256) void reshape_qkv(const __bf16* __restrict__ qkv,
    __bf16* __restrict__ q, __bf16* __restrict__ k, __bf16* __restrict__ vt)
{
  size_t i = (size_t)blockIdx.x * blockDim.x + threadIdx.x;   // over BT*512
  if (i >= (size_t)BT * CH) return;
  int col = (int)(i & (CH - 1));          // h*64+d
  size_t rowi = i >> 9;                   // b*361+t
  int b = (int)(rowi / TOK), t = (int)(rowi % TOK);
  int h = col >> 6, d = col & 63;
  size_t base = rowi * (3 * CH);
  size_t bh = (size_t)b * NH + h;
  float qv = (float)qkv[base + col] * 0.125f;           // DH^-0.5 = 1/8
  q[(bh * TPAD + t) * DHEAD + d]  = (__bf16)qv;
  k[(bh * TPAD + t) * DHEAD + d]  = qkv[base + CH + col];
  vt[(bh * DHEAD + d) * TPAD + t] = qkv[base + 2 * CH + col];
}

// ---------------- attention: wave per (b,h, 16-row tile) ---------------------
__global__ __launch_bounds__(32) void attn_kernel(
    const __bf16* __restrict__ qb, const __bf16* __restrict__ kbuf,
    const __bf16* __restrict__ vtb, const float* __restrict__ relb,
    __bf16* __restrict__ ctx)
{
  __shared__ float lg[16][TPAD];          // 24 KB logits/probs staging
  int bh = blockIdx.x / 23;
  int tt = blockIdx.x % 23;
  int b = bh >> 3, h = bh & 7;
  int t0 = tt * 16;
  int lane = threadIdx.x;
  int ln15 = lane & 15;
  int kb8  = (lane >> 4) * 8;
  int kb16 = (lane >> 4) * 16;

  // q A-fragments (K = head dim, 2 fragments of 32)
  const __bf16* qrow = qb + ((size_t)bh * TPAD + t0 + ln15) * DHEAD;
  Frag qa0, qa1;
  qa0.q[0] = *(const uint4*)(qrow + kb8);
  qa0.q[1] = *(const uint4*)(qrow + kb8 + 16);
  qa1.q[0] = *(const uint4*)(qrow + 32 + kb8);
  qa1.q[1] = *(const uint4*)(qrow + 48 + kb8);

  // logits = q k^T + rel_bias, staged to LDS
  for (int st = 0; st < 24; ++st) {
    int s0 = st * 16;
    const __bf16* krow = kbuf + ((size_t)bh * TPAD + s0 + ln15) * DHEAD;
    Frag kb0, kb1;
    kb0.q[0] = *(const uint4*)(krow + kb8);
    kb0.q[1] = *(const uint4*)(krow + kb8 + 16);
    kb1.q[0] = *(const uint4*)(krow + 32 + kb8);
    kb1.q[1] = *(const uint4*)(krow + 48 + kb8);
    v8f c = {};
    c = wmma_bf16(qa0.v, kb0.v, c);
    c = wmma_bf16(qa1.v, kb1.v, c);
    int s = s0 + ln15;
    int rs = s < TOK ? s / 19 : 0;
    int cs = s < TOK ? s % 19 : 0;
#pragma unroll
    for (int i = 0; i < 8; ++i) {
      int m = (lane >> 4) * 8 + i;
      int t = t0 + m;
      float val = -1e30f;
      if (s < TOK && t < TOK) {
        int rt = t / 19, ct = t % 19;
        int idx = (rt - rs + 18) * 37 + (ct - cs + 18);
        val = c[i] + relb[h * 1369 + idx];
      }
      lg[m][s] = val;
    }
  }
  __syncthreads();

  // in-wave softmax over each of the 16 rows
  for (int r = 0; r < 16; ++r) {
    float mx = -1e30f;
    for (int j = lane; j < TPAD; j += 32) mx = fmaxf(mx, lg[r][j]);
#pragma unroll
    for (int off = 16; off; off >>= 1) mx = fmaxf(mx, __shfl_xor(mx, off, 32));
    float sum = 0.f;
    for (int j = lane; j < TPAD; j += 32) {
      float e = __expf(lg[r][j] - mx);
      lg[r][j] = e;
      sum += e;
    }
#pragma unroll
    for (int off = 16; off; off >>= 1) sum += __shfl_xor(sum, off, 32);
    float inv = 1.f / sum;
    for (int j = lane; j < TPAD; j += 32) lg[r][j] *= inv;
  }
  __syncthreads();

  // ctx = probs @ v   (probs A-fragments built from LDS, vT gives contiguous B)
  v8f cacc[4] = {};
  int r = ln15;
  for (int kc = 0; kc < TPAD; kc += 32) {
    v16bf av;
#pragma unroll
    for (int j = 0; j < 8; ++j) {
      av[j]     = (__bf16)lg[r][kc + kb8 + j];
      av[j + 8] = (__bf16)lg[r][kc + 16 + kb8 + j];
    }
#pragma unroll
    for (int ntile = 0; ntile < 4; ++ntile) {
      const __bf16* vrow = vtb + ((size_t)bh * DHEAD + ntile * 16 + ln15) * TPAD + kc + kb8;
      Frag bf;
      bf.q[0] = *(const uint4*)(vrow);
      bf.q[1] = *(const uint4*)(vrow + 16);
      cacc[ntile] = wmma_bf16(av, bf.v, cacc[ntile]);
    }
  }
#pragma unroll
  for (int ntile = 0; ntile < 4; ++ntile)
#pragma unroll
    for (int i = 0; i < 8; ++i) {
      int t = t0 + (lane >> 4) * 8 + i;
      if (t < TOK) {
        int d = ntile * 16 + ln15;
        ctx[((size_t)b * TOK + t) * CH + h * DHEAD + d] = (__bf16)cacc[ntile][i];
      }
    }
}

// ----------------------------------------------------------------------------
extern "C" void kernel_launch(void* const* d_in, const int* in_sizes, int n_in,
                              void* d_out, int out_size, void* d_ws, size_t ws_size,
                              hipStream_t stream) {
  const float* x     = (const float*)d_in[0];
  const float* ln1_s = (const float*)d_in[1];
  const float* ln1_b = (const float*)d_in[2];
  const float* Wq    = (const float*)d_in[3];
  const float* bq    = (const float*)d_in[4];
  const float* Wk    = (const float*)d_in[5];
  const float* bk    = (const float*)d_in[6];
  const float* Wv    = (const float*)d_in[7];
  const float* bv    = (const float*)d_in[8];
  const float* Wo    = (const float*)d_in[9];
  const float* bo    = (const float*)d_in[10];
  const float* relb  = (const float*)d_in[11];
  const float* ln2_s = (const float*)d_in[12];
  const float* ln2_b = (const float*)d_in[13];
  const float* W1    = (const float*)d_in[14];
  const float* b1    = (const float*)d_in[15];
  const float* W2    = (const float*)d_in[16];
  const float* b2    = (const float*)d_in[17];

  char* ws = (char*)d_ws;
  size_t off = 0;
  auto alloc = [&](size_t bytes) -> void* {
    off = (off + 255) & ~(size_t)255;
    void* p = ws + off;
    off += bytes;
    return p;
  };

  __bf16* WqkvT = (__bf16*)alloc((size_t)1536 * 512 * 2);
  __bf16* WoT   = (__bf16*)alloc((size_t)512 * 512 * 2);
  __bf16* W1T   = (__bf16*)alloc((size_t)2048 * 512 * 2);
  __bf16* W2T   = (__bf16*)alloc((size_t)512 * 2048 * 2);
  float*  bqkv  = (float*) alloc(1536 * 4);
  __bf16* ybuf  = (__bf16*)alloc((size_t)BT * CH * 2);          // LN1 out, reused for LN2 out
  __bf16* qkv   = (__bf16*)alloc((size_t)BT * 3 * CH * 2);      // GEMM1 out (dead after reshape)
  __bf16* qbuf  = (__bf16*)alloc((size_t)BH * TPAD * DHEAD * 2);
  __bf16* kbuf  = (__bf16*)alloc((size_t)BH * TPAD * DHEAD * 2);
  __bf16* vtbuf = (__bf16*)alloc((size_t)BH * DHEAD * TPAD * 2);
  __bf16* ctx   = (__bf16*)alloc((size_t)BT * CH * 2);
  float*  xattn = (float*) alloc((size_t)BT * CH * 4);
  __bf16* hbuf  = qkv;  // MLP hidden [BT,2048] bf16 aliases qkv+qbuf (both dead by then)

  // 1) weights -> bf16 transposed
  cvt_w<<<(512 * 512 + 255) / 256, 256, 0, stream>>>(Wq, WqkvT,                 512, 512);
  cvt_w<<<(512 * 512 + 255) / 256, 256, 0, stream>>>(Wk, WqkvT + 512 * 512,     512, 512);
  cvt_w<<<(512 * 512 + 255) / 256, 256, 0, stream>>>(Wv, WqkvT + 1024 * 512,    512, 512);
  cvt_w<<<(512 * 512 + 255) / 256, 256, 0, stream>>>(Wo, WoT,                   512, 512);
  cvt_w<<<(512 * 2048 + 255) / 256, 256, 0, stream>>>(W1, W1T,                  512, 2048);
  cvt_w<<<(2048 * 512 + 255) / 256, 256, 0, stream>>>(W2, W2T,                  2048, 512);
  pack_bias<<<6, 256, 0, stream>>>(bq, bk, bv, bqkv);

  // 2) LN1
  ln_kernel<<<(BT + 7) / 8, 256, 0, stream>>>(x, ln1_s, ln1_b, ybuf, BT);

  // 3) QKV projection: [BT,512] x [512,1536] -> bf16  (grid exact: waves % 8 == 0)
  {
    int waves = (BT / 32) * (1536 / 64);
    gemm_bf16<<<waves / 8, 256, 0, stream>>>(ybuf, WqkvT, bqkv, nullptr,
                                             nullptr, qkv, BT, 1536, 512, 0);
  }

  // 4) scatter to per-head padded layouts (q pre-scaled by DH^-0.5)
  reshape_qkv<<<(int)(((size_t)BT * CH + 255) / 256), 256, 0, stream>>>(qkv, qbuf, kbuf, vtbuf);

  // 5) attention
  attn_kernel<<<BH * 23, 32, 0, stream>>>(qbuf, kbuf, vtbuf, relb, ctx);

  // 6) output projection + residual -> xattn f32
  {
    int waves = (BT / 32) * (512 / 64);
    gemm_bf16<<<waves / 8, 256, 0, stream>>>(ctx, WoT, bo, x,
                                             xattn, nullptr, BT, 512, 512, 0);
  }

  // 7) LN2
  ln_kernel<<<(BT + 7) / 8, 256, 0, stream>>>(xattn, ln2_s, ln2_b, ybuf, BT);

  // 8) MLP up + GELU -> bf16
  {
    int waves = (BT / 32) * (2048 / 64);
    gemm_bf16<<<waves / 8, 256, 0, stream>>>(ybuf, W1T, b1, nullptr,
                                             nullptr, hbuf, BT, 2048, 512, 1);
  }

  // 9) MLP down + residual -> d_out f32
  {
    int waves = (BT / 32) * (512 / 64);
    gemm_bf16<<<waves / 8, 256, 0, stream>>>(hbuf, W2T, b2, xattn,
                                             (float*)d_out, nullptr, BT, 512, 2048, 0);
  }
  (void)in_sizes; (void)n_in; (void)out_size; (void)ws_size;
}
